// MambaEncoder_42863773614418
// MI455X (gfx1250) — matrix operand order repaired and checked by reading, not compiled, not run on hardware
//
#include <hip/hip_runtime.h>
#include <hip/hip_bf16.h>
#include <math.h>

// ---------------------------------------------------------------------------
// Problem constants (from reference)
// ---------------------------------------------------------------------------
#define BB     8
#define LL     1024
#define M_ROWS (BB * LL)      // 8192 token rows
#define IN_D   8
#define DOUT   128
#define DIN    256            // 2*DOUT
#define DSTATE 32
#define DCONV  4
#define DTR    8
#define NPROJ  72             // DTR + 2*DSTATE
#define NPAD   80             // NPROJ padded to multiple of 16

typedef __attribute__((ext_vector_type(16))) _Float16 v16h;
typedef __attribute__((ext_vector_type(8)))  _Float16 v8h;
typedef __attribute__((ext_vector_type(8)))  float    v8f;

// ---------------------------------------------------------------------------
// f32 -> f16 conversion helpers
// ---------------------------------------------------------------------------
__global__ __launch_bounds__(256) void cvt_f32_to_f16(const float* __restrict__ src,
                                                      _Float16* __restrict__ dst, int n) {
    int t = blockIdx.x * blockDim.x + threadIdx.x;
    if (t < n) dst[t] = (_Float16)src[t];
}

// Convert [rows x cols] into [padRows x cols], zero-filling padded rows.
__global__ __launch_bounds__(256) void cvt_f32_to_f16_padrows(const float* __restrict__ src,
                                                              _Float16* __restrict__ dst,
                                                              int rows, int cols, int padRows) {
    int t = blockIdx.x * blockDim.x + threadIdx.x;
    int n = padRows * cols;
    if (t >= n) return;
    int r = t / cols;
    dst[t] = (r < rows) ? (_Float16)src[t] : (_Float16)0.0f;
}

// ---------------------------------------------------------------------------
// WMMA f16 GEMM:  C[M,N] = A[M,K] @ W[N,K]^T (+ bias[N])
//   A row-major f16, W row-major [N][K] f16 (native Linear layout).
//   2x2 register blocking: each wave computes a 32x32 output patch with
//   4 fp32 accumulators; per K-step 2 A-frags + 2 B-frags feed 4 WMMAs
//   (2 b128 loads per v_wmma_f32_16x16x32_f16 instead of 4).
//   4 waves / 128-thread block, waves strided over n-tile pairs.
// ---------------------------------------------------------------------------
__global__ __launch_bounds__(128) void gemm_f16_wmma(const _Float16* __restrict__ A,
                                                     const _Float16* __restrict__ W,
                                                     const float* __restrict__ bias,
                                                     float* __restrict__ C,
                                                     int M, int N, int K) {
    const int lane = threadIdx.x & 31;
    const int wid  = threadIdx.x >> 5;
    const int ntp  = blockIdx.x * 4 + wid;     // n-tile pair index
    const int nt0  = ntp * 2;
    if (nt0 * 16 >= N) return;
    const int  nt1   = nt0 + 1;
    const bool hasN1 = (nt1 * 16 < N);
    const int  nt1c  = hasN1 ? nt1 : nt0;      // clamp loads if odd tile count
    const int  mt0   = blockIdx.y * 2;
    const int  mt1   = mt0 + 1;

    const int half = lane >> 4;                // 0: lanes 0-15, 1: lanes 16-31
    const int mr   = lane & 15;
    const int m0 = mt0 * 16 + mr, m1 = mt1 * 16 + mr;   // A rows held by lane
    const int n0 = nt0 * 16 + mr, n1 = nt1c * 16 + mr;  // W rows / out cols

    // A 16x32 f16 layout: lane m, elems 0..15 = K[half*16 .. half*16+15]
    // B 32x16 f16 layout: lane n, elems 0..15 = K[half*16 .. half*16+15]
    const _Float16* a0p = A + (size_t)m0 * K + half * 16;
    const _Float16* a1p = A + (size_t)m1 * K + half * 16;
    const _Float16* b0p = W + (size_t)n0 * K + half * 16;
    const _Float16* b1p = W + (size_t)n1 * K + half * 16;

    v8f c00 = {}, c01 = {}, c10 = {}, c11 = {};
    for (int k0 = 0; k0 < K; k0 += 32) {
        v8h lo, hi;
        v16h a0, a1, b0, b1;
        lo = *(const v8h*)(a0p + k0); hi = *(const v8h*)(a0p + k0 + 8);
#pragma unroll
        for (int i = 0; i < 8; ++i) { a0[i] = lo[i]; a0[i + 8] = hi[i]; }
        lo = *(const v8h*)(a1p + k0); hi = *(const v8h*)(a1p + k0 + 8);
#pragma unroll
        for (int i = 0; i < 8; ++i) { a1[i] = lo[i]; a1[i + 8] = hi[i]; }
        b0 = *(const v16h*)(b0p + k0);
        b1 = *(const v16h*)(b1p + k0);
        c00 = __builtin_amdgcn_wmma_f32_16x16x32_f16(false, a0, false, b0, (short)0, c00, false, false);
        c01 = __builtin_amdgcn_wmma_f32_16x16x32_f16(false, a0, false, b1, (short)0, c01, false, false);
        c10 = __builtin_amdgcn_wmma_f32_16x16x32_f16(false, a1, false, b0, (short)0, c10, false, false);
        c11 = __builtin_amdgcn_wmma_f32_16x16x32_f16(false, a1, false, b1, (short)0, c11, false, false);
    }
    const float bn0 = bias ? bias[n0] : 0.0f;
    const float bn1 = bias ? bias[n1] : 0.0f;
#pragma unroll
    for (int r = 0; r < 8; ++r) {
        // C/D layout: VGPR r -> M = half*8 + r, N = lane&15
        const int r0 = mt0 * 16 + half * 8 + r;
        const int r1 = mt1 * 16 + half * 8 + r;
        C[(size_t)r0 * N + n0] = c00[r] + bn0;
        C[(size_t)r1 * N + n0] = c10[r] + bn0;
        if (hasN1) {
            C[(size_t)r0 * N + n1] = c01[r] + bn1;
            C[(size_t)r1 * N + n1] = c11[r] + bn1;
        }
    }
}

// ---------------------------------------------------------------------------
// Front-end: 8 -> 16 -> 32 -> 128 MLP fused with DyTanh1. One block per token.
// ---------------------------------------------------------------------------
__global__ __launch_bounds__(128) void proj_dytanh_kernel(
    const float* __restrict__ x,
    const float* __restrict__ W1, const float* __restrict__ b1,
    const float* __restrict__ W2, const float* __restrict__ b2,
    const float* __restrict__ W3, const float* __restrict__ b3,
    const float* __restrict__ alpha, const float* __restrict__ dyw,
    const float* __restrict__ dyb, float* __restrict__ H) {
    __shared__ float xs[IN_D];
    __shared__ float t1[16];
    __shared__ float t2[32];
    const int p = blockIdx.x;          // token index 0..8191
    const int tid = threadIdx.x;       // 0..127
    if (tid < IN_D) xs[tid] = x[(size_t)p * IN_D + tid];
    __syncthreads();
    if (tid < 16) {
        float s = b1[tid];
#pragma unroll
        for (int j = 0; j < IN_D; ++j) s += W1[tid * IN_D + j] * xs[j];
        t1[tid] = s;
    }
    __syncthreads();
    if (tid < 32) {
        float s = b2[tid];
#pragma unroll
        for (int j = 0; j < 16; ++j) s += W2[tid * 16 + j] * t1[j];
        t2[tid] = s;
    }
    __syncthreads();
    float s = b3[tid];
#pragma unroll
    for (int j = 0; j < 32; ++j) s += W3[tid * 32 + j] * t2[j];
    H[(size_t)p * DOUT + tid] = dyw[tid] * tanhf(alpha[0] * s) + dyb[tid];
}

// ---------------------------------------------------------------------------
// Depthwise causal conv1d (DCONV=4) over time + SiLU.  xi = xz[:, :DIN].
// ---------------------------------------------------------------------------
__global__ __launch_bounds__(256) void conv_silu_kernel(const float* __restrict__ xz,
                                                        const float* __restrict__ convw,
                                                        const float* __restrict__ convb,
                                                        float* __restrict__ xc) {
    int t = blockIdx.x * blockDim.x + threadIdx.x;
    if (t >= M_ROWS * DIN) return;
    const int d = t & (DIN - 1);
    const int l = (t >> 8) & (LL - 1);
    const int b = t >> 18;
    const float* w = convw + d * DCONV;
    float s = convb[d];
#pragma unroll
    for (int j = 0; j < DCONV; ++j) {
        int ls = l - (DCONV - 1) + j;
        if (ls >= 0) s += w[j] * xz[((size_t)(b * LL + ls)) * (2 * DIN) + d];
    }
    xc[t] = s / (1.0f + __expf(-s));   // silu
}

// ---------------------------------------------------------------------------
// dt = softplus(dbl[:, :DTR] @ Wdt^T + bdt)
// ---------------------------------------------------------------------------
__global__ __launch_bounds__(256) void dt_softplus_kernel(const float* __restrict__ dbl,
                                                          const float* __restrict__ Wdt,
                                                          const float* __restrict__ bdt,
                                                          float* __restrict__ DT) {
    int t = blockIdx.x * blockDim.x + threadIdx.x;
    if (t >= M_ROWS * DIN) return;
    const int d = t & (DIN - 1);
    const int pos = t >> 8;            // b*L + l
    float s = bdt[d];
    const float* row = dbl + (size_t)pos * NPAD;
#pragma unroll
    for (int r = 0; r < DTR; ++r) s += Wdt[d * DTR + r] * row[r];
    DT[t] = (s > 20.0f) ? s : log1pf(__expf(s));
}

// ---------------------------------------------------------------------------
// Selective scan + gating, fused.  One wave32 per (batch, channel) pair;
// lane = state index n (DSTATE == 32 == wave width).  Cross-lane C-dot via
// xor-shuffle reduction.  Prefetches ~8 steps ahead (global_prefetch_b8)
// to hide L2 latency on the serial recurrence (the true critical path).
// Writes YS = (scan + xc*D) * silu(z).
// ---------------------------------------------------------------------------
__global__ __launch_bounds__(256) void scan_gate_kernel(const float* __restrict__ DT,
                                                        const float* __restrict__ XC,
                                                        const float* __restrict__ XZ,
                                                        const float* __restrict__ DBL,
                                                        const float* __restrict__ Alog,
                                                        const float* __restrict__ Dp,
                                                        float* __restrict__ YS) {
    const int wave = (blockIdx.x * blockDim.x + threadIdx.x) >> 5;  // 0..2047
    const int lane = threadIdx.x & 31;                              // state n
    const int b = wave >> 8;            // wave / DIN
    const int d = wave & (DIN - 1);
    const float An = -__expf(Alog[d * DSTATE + lane]);
    const float Dd = Dp[d];
    float h = 0.0f;
    const size_t base = (size_t)b * LL;
    for (int l = 0; l < LL; ++l) {
        const size_t pos = base + l;
        if (l + 8 < LL) {
            const size_t pf = pos + 8;
            __builtin_prefetch(&DBL[pf * NPAD + DTR + lane], 0, 1);
            __builtin_prefetch(&DT[pf * DIN + d], 0, 1);
            __builtin_prefetch(&XC[pf * DIN + d], 0, 1);
        }
        const float dt = DT[pos * DIN + d];
        const float xc = XC[pos * DIN + d];
        const float Bv = DBL[pos * NPAD + DTR + lane];
        const float Cv = DBL[pos * NPAD + DTR + DSTATE + lane];
        const float dA = __expf(dt * An);
        h = dA * h + (dt * xc) * Bv;
        float p = h * Cv;
#pragma unroll
        for (int off = 16; off > 0; off >>= 1) p += __shfl_xor(p, off, 32);
        if (lane == 0) {
            const float z = XZ[pos * (2 * DIN) + DIN + d];
            const float sz = z / (1.0f + __expf(-z));
            YS[pos * DIN + d] = (p + xc * Dd) * sz;
        }
    }
}

// ---------------------------------------------------------------------------
// Final DyTanh with per-(l,d) scale/shift.
// ---------------------------------------------------------------------------
__global__ __launch_bounds__(256) void dytanh2_kernel(const float* __restrict__ H,
                                                      const float* __restrict__ alpha,
                                                      const float* __restrict__ w,
                                                      const float* __restrict__ bshift,
                                                      float* __restrict__ out) {
    int t = blockIdx.x * blockDim.x + threadIdx.x;
    if (t >= M_ROWS * DOUT) return;
    const int wi = t & (LL * DOUT - 1);   // l*128 + d
    out[t] = w[wi] * tanhf(alpha[0] * H[t]) + bshift[wi];
}

// ---------------------------------------------------------------------------
// Orchestration
// ---------------------------------------------------------------------------
extern "C" void kernel_launch(void* const* d_in, const int* in_sizes, int n_in,
                              void* d_out, int out_size, void* d_ws, size_t ws_size,
                              hipStream_t stream) {
    (void)in_sizes; (void)n_in; (void)out_size; (void)ws_size;

    const float* x  = (const float*)d_in[0];
    const float* W1 = (const float*)d_in[1];  const float* b1 = (const float*)d_in[2];
    const float* W2 = (const float*)d_in[3];  const float* b2 = (const float*)d_in[4];
    const float* W3 = (const float*)d_in[5];  const float* b3 = (const float*)d_in[6];
    const float* dy1_a = (const float*)d_in[7];
    const float* dy1_w = (const float*)d_in[8];
    const float* dy1_b = (const float*)d_in[9];
    const float* dy2_a = (const float*)d_in[32];
    const float* dy2_w = (const float*)d_in[33];
    const float* dy2_b = (const float*)d_in[34];

    // Workspace layout (floats unless noted)
    float* H   = (float*)d_ws;                       // 8192*128
    float* XZ  = H   + (size_t)M_ROWS * DOUT;        // 8192*512
    float* XC  = XZ  + (size_t)M_ROWS * 2 * DIN;     // 8192*256
    float* DBL = XC  + (size_t)M_ROWS * DIN;         // 8192*80
    float* DT  = DBL + (size_t)M_ROWS * NPAD;        // 8192*256
    float* YS  = DT  + (size_t)M_ROWS * DIN;         // 8192*256
    _Float16* AH = (_Float16*)(YS + (size_t)M_ROWS * DIN);  // 8192*256 halves
    _Float16* WH = AH + (size_t)M_ROWS * DIN;               // up to 512*128 halves

    // grid helper: waves cover n-tile pairs (32 cols), blocks cover 32 rows
    auto gemm_grid = [](int N) {
        int pairs = (N / 16 + 1) / 2;
        return dim3((pairs + 3) / 4, M_ROWS / 32);
    };

    // 1) Front-end MLP + DyTanh1
    hipLaunchKernelGGL(proj_dytanh_kernel, dim3(M_ROWS), dim3(DOUT), 0, stream,
                       x, W1, b1, W2, b2, W3, b3, dy1_a, dy1_w, dy1_b, H);

    // 2) Two Mamba blocks
    for (int blk = 0; blk < 2; ++blk) {
        const int base = 10 + blk * 11;
        const float* Win   = (const float*)d_in[base + 0];
        const float* bin   = (const float*)d_in[base + 1];
        const float* convw = (const float*)d_in[base + 2];
        const float* convb = (const float*)d_in[base + 3];
        const float* Wxp   = (const float*)d_in[base + 4];
        const float* Wdt   = (const float*)d_in[base + 5];
        const float* bdt   = (const float*)d_in[base + 6];
        const float* Alog  = (const float*)d_in[base + 7];
        const float* Dp    = (const float*)d_in[base + 8];
        const float* Wout  = (const float*)d_in[base + 9];
        const float* bout  = (const float*)d_in[base + 10];

        // xz = H @ Win^T + bin   (M=8192, N=512, K=128)  -- WMMA
        hipLaunchKernelGGL(cvt_f32_to_f16, dim3((M_ROWS * DOUT + 255) / 256), dim3(256), 0, stream,
                           H, AH, M_ROWS * DOUT);
        hipLaunchKernelGGL(cvt_f32_to_f16, dim3((2 * DIN * DOUT + 255) / 256), dim3(256), 0, stream,
                           Win, WH, 2 * DIN * DOUT);
        hipLaunchKernelGGL(gemm_f16_wmma, gemm_grid(2 * DIN), dim3(128), 0, stream,
                           AH, WH, bin, XZ, M_ROWS, 2 * DIN, DOUT);

        // depthwise causal conv + silu
        hipLaunchKernelGGL(conv_silu_kernel, dim3((M_ROWS * DIN + 255) / 256), dim3(256), 0, stream,
                           XZ, convw, convb, XC);

        // dbl = xc @ Wxp^T       (M=8192, N=80 padded, K=256)  -- WMMA
        hipLaunchKernelGGL(cvt_f32_to_f16, dim3((M_ROWS * DIN + 255) / 256), dim3(256), 0, stream,
                           XC, AH, M_ROWS * DIN);
        hipLaunchKernelGGL(cvt_f32_to_f16_padrows, dim3((NPAD * DIN + 255) / 256), dim3(256), 0, stream,
                           Wxp, WH, NPROJ, DIN, NPAD);
        hipLaunchKernelGGL(gemm_f16_wmma, gemm_grid(NPAD), dim3(128), 0, stream,
                           AH, WH, (const float*)nullptr, DBL, M_ROWS, NPAD, DIN);

        // dt = softplus(dbl[:, :8] @ Wdt^T + bdt)
        hipLaunchKernelGGL(dt_softplus_kernel, dim3((M_ROWS * DIN + 255) / 256), dim3(256), 0, stream,
                           DBL, Wdt, bdt, DT);

        // selective scan + gating (wave per (b,d), lane = state)
        hipLaunchKernelGGL(scan_gate_kernel, dim3(BB * DIN * 32 / 256), dim3(256), 0, stream,
                           DT, XC, XZ, DBL, Alog, Dp, YS);

        // H = ys @ Wout^T + bout (M=8192, N=128, K=256)  -- WMMA
        hipLaunchKernelGGL(cvt_f32_to_f16, dim3((M_ROWS * DIN + 255) / 256), dim3(256), 0, stream,
                           YS, AH, M_ROWS * DIN);
        hipLaunchKernelGGL(cvt_f32_to_f16, dim3((DOUT * DIN + 255) / 256), dim3(256), 0, stream,
                           Wout, WH, DOUT * DIN);
        hipLaunchKernelGGL(gemm_f16_wmma, gemm_grid(DOUT), dim3(128), 0, stream,
                           AH, WH, bout, H, M_ROWS, DOUT, DIN);
    }

    // 3) Final per-position DyTanh
    hipLaunchKernelGGL(dytanh2_kernel, dim3((M_ROWS * DOUT + 255) / 256), dim3(256), 0, stream,
                       H, dy2_a, dy2_w, dy2_b, (float*)d_out);
}
